// GraphQNetwork_9242769622574
// MI455X (gfx1250) — matrix operand order repaired and checked by reading, not compiled, not run on hardware
//
#include <hip/hip_runtime.h>
#include <stdint.h>

typedef __attribute__((ext_vector_type(16))) __bf16 v16bf;
typedef __attribute__((ext_vector_type(8)))  float  v8f;

#define N_NODES 32768
#define N_PER   512
#define NGRAPH  64
#define NEDGE   524288
#define NHEADS  4
#define E2      (NEDGE + N_NODES)   // edges + self loops

__constant__ int c_stations[16] = {0,31,63,95,127,159,191,223,255,287,319,351,383,415,447,479};

// ---------- helpers ----------
__device__ __forceinline__ unsigned short f2bf(float f) {
  unsigned b = __float_as_uint(f);
  unsigned r = b + 0x7fffu + ((b >> 16) & 1u);   // round-to-nearest-even
  return (unsigned short)(r >> 16);
}
__device__ __forceinline__ unsigned f2o(float f) {   // order-preserving float->uint
  unsigned b = __float_as_uint(f);
  return (b & 0x80000000u) ? ~b : (b | 0x80000000u);
}
__device__ __forceinline__ float o2f(unsigned o) {
  unsigned b = (o & 0x80000000u) ? (o & 0x7fffffffu) : ~o;
  return __uint_as_float(b);
}

// ---------- generic fills / converts ----------
__global__ void fill_u32(unsigned* p, unsigned v, int n) {
  int i = blockIdx.x * blockDim.x + threadIdx.x;
  if (i < n) p[i] = v;
}
__global__ void cvt_f32_bf16(const float* in, unsigned short* out, int n) {
  int i = blockIdx.x * blockDim.x + threadIdx.x;
  if (i < n) out[i] = f2bf(in[i]);
}
// W[K,Nc] row-major f32 -> Wt[Nc,K] bf16 (transposed, so B fragments load contiguously)
__global__ void repack_wt(const float* W, unsigned short* Wt, int K, int Nc) {
  int i = blockIdx.x * blockDim.x + threadIdx.x;
  if (i >= K * Nc) return;
  int k = i / Nc, n = i % Nc;
  Wt[(size_t)n * K + k] = f2bf(W[i]);
}

// ---------- WMMA GEMM: C[M,Nc] = A[M,K](bf16) * W[K,Nc] + bias ----------
// A row-major bf16, Wt = W transposed [Nc,K] bf16.
// One 16(M) x 64(N) strip per wave: A fragment loaded once per k-step, 4 WMMAs.
__global__ __launch_bounds__(256) void gemm_wmma_bf16(
    const unsigned short* __restrict__ A, const unsigned short* __restrict__ Wt,
    const float* __restrict__ bias, float* __restrict__ C, int M, int K, int Nc) {
  int wave = threadIdx.x >> 5;
  int lane = threadIdx.x & 31;
  int strip = blockIdx.x * 8 + wave;     // each strip = 16 rows x 64 cols
  int stripsN = Nc >> 6;                 // Nc / 64
  int tm = strip / stripsN, tg = strip % stripsN;
  if (tm * 16 >= M) return;              // uniform per wave
  int lo = lane & 15, hi = lane >> 4;

  const unsigned* Arow = (const unsigned*)(A + (size_t)(tm * 16 + lo) * K);
  const unsigned* Brow0 = (const unsigned*)(Wt + (size_t)(tg * 64 + 0 * 16 + lo) * K);
  const unsigned* Brow1 = (const unsigned*)(Wt + (size_t)(tg * 64 + 1 * 16 + lo) * K);
  const unsigned* Brow2 = (const unsigned*)(Wt + (size_t)(tg * 64 + 2 * 16 + lo) * K);
  const unsigned* Brow3 = (const unsigned*)(Wt + (size_t)(tg * 64 + 3 * 16 + lo) * K);

  v8f acc0 = {0.f,0.f,0.f,0.f,0.f,0.f,0.f,0.f};
  v8f acc1 = acc0, acc2 = acc0, acc3 = acc0;

  for (int k0 = 0; k0 < K; k0 += 32) {
    union { v16bf v; unsigned u[8]; } au, b0, b1, b2, b3;
    const unsigned* Ap  = Arow  + (k0 >> 1);
    const unsigned* Bp0 = Brow0 + (k0 >> 1);
    const unsigned* Bp1 = Brow1 + (k0 >> 1);
    const unsigned* Bp2 = Brow2 + (k0 >> 1);
    const unsigned* Bp3 = Brow3 + (k0 >> 1);
#pragma unroll
    for (int j = 0; j < 8; ++j) {
      // A 16x32 bf16 fragment: lane lo holds M=lo; k = hi*16 + 2*j + p
      au.u[j] = Ap[hi * 8 + j];
      // B 32x16 bf16 fragment: lane lo = column; k = hi*16 + 2*j + p
      int bj = hi * 8 + j;
      b0.u[j] = Bp0[bj];
      b1.u[j] = Bp1[bj];
      b2.u[j] = Bp2[bj];
      b3.u[j] = Bp3[bj];
    }
    acc0 = __builtin_amdgcn_wmma_f32_16x16x32_bf16(false, au.v, false, b0.v, (short)0, acc0, false, false);
    acc1 = __builtin_amdgcn_wmma_f32_16x16x32_bf16(false, au.v, false, b1.v, (short)0, acc1, false, false);
    acc2 = __builtin_amdgcn_wmma_f32_16x16x32_bf16(false, au.v, false, b2.v, (short)0, acc2, false, false);
    acc3 = __builtin_amdgcn_wmma_f32_16x16x32_bf16(false, au.v, false, b3.v, (short)0, acc3, false, false);
  }

#pragma unroll
  for (int t = 0; t < 4; ++t) {
    int col = tg * 64 + t * 16 + lo;
    float bv = bias[col];
    v8f acc = (t == 0) ? acc0 : (t == 1) ? acc1 : (t == 2) ? acc2 : acc3;
#pragma unroll
    for (int r = 0; r < 8; ++r)
      C[(size_t)(tm * 16 + hi * 8 + r) * Nc + col] = acc[r] + bv;
  }
}

// ---------- degree + self-loop edge attr (fill_value='mean') ----------
__global__ void deg_loop_acc(const int* __restrict__ dst, const float* __restrict__ eattr,
                             float* cnt, float* lsum) {
  int e = blockIdx.x * blockDim.x + threadIdx.x;
  if (e >= NEDGE) return;
  int d = dst[e];
  atomicAdd(cnt + d, 1.f);
  atomicAdd(lsum + 2 * d,     eattr[2 * e]);
  atomicAdd(lsum + 2 * d + 1, eattr[2 * e + 1]);
}
__global__ void loop_fin(const float* cnt, const float* lsum, float* lattr) {
  int n = blockIdx.x * blockDim.x + threadIdx.x;
  if (n >= N_NODES) return;
  float c = fmaxf(cnt[n], 1.f);
  lattr[2 * n]     = lsum[2 * n] / c;
  lattr[2 * n + 1] = lsum[2 * n + 1] / c;
}

// ---------- pass 1: edge scores + per-(dst,head) running max ----------
__global__ void edge_score_max(const int* __restrict__ src, const int* __restrict__ dst,
                               const float* __restrict__ eattr, const float* __restrict__ lattr,
                               const float* __restrict__ xl, const float* __restrict__ xr,
                               const float* __restrict__ We, const float* __restrict__ att,
                               float* __restrict__ score, unsigned* __restrict__ smax, int C) {
  int idx = blockIdx.x * blockDim.x + threadIdx.x;
  if (idx >= E2 * NHEADS) return;
  int h = idx & 3, e = idx >> 2;
  int s, d; float a0, a1;
  if (e < NEDGE) { s = src[e]; d = dst[e]; a0 = eattr[2 * e]; a1 = eattr[2 * e + 1]; }
  else { s = d = e - NEDGE; a0 = lattr[2 * (e - NEDGE)]; a1 = lattr[2 * (e - NEDGE) + 1]; }
  int HC = NHEADS * C, base = h * C;
  const float4* pl = (const float4*)(xl + (size_t)s * HC + base);
  const float4* pr = (const float4*)(xr + (size_t)d * HC + base);
  const float4* w0 = (const float4*)(We + base);
  const float4* w1 = (const float4*)(We + HC + base);
  const float4* pa = (const float4*)(att + base);
  float sc = 0.f;
  for (int c4 = 0; c4 < (C >> 2); ++c4) {
    float4 l = pl[c4], r = pr[c4], e0 = w0[c4], e1 = w1[c4], at = pa[c4];
    float m;
    m = l.x + r.x + a0 * e0.x + a1 * e1.x; m = m > 0.f ? m : 0.2f * m; sc += m * at.x;
    m = l.y + r.y + a0 * e0.y + a1 * e1.y; m = m > 0.f ? m : 0.2f * m; sc += m * at.y;
    m = l.z + r.z + a0 * e0.z + a1 * e1.z; m = m > 0.f ? m : 0.2f * m; sc += m * at.z;
    m = l.w + r.w + a0 * e0.w + a1 * e1.w; m = m > 0.f ? m : 0.2f * m; sc += m * at.w;
  }
  score[idx] = sc;
  atomicMax(smax + (size_t)d * NHEADS + h, f2o(sc));
}

// ---------- pass 2: exp, denom, numerator scatter-add ----------
__global__ void edge_accum(const int* __restrict__ src, const int* __restrict__ dst,
                           const float* __restrict__ score, const unsigned* __restrict__ smax,
                           const float* __restrict__ xl, float* __restrict__ denom,
                           float* __restrict__ accum, int C) {
  int idx = blockIdx.x * blockDim.x + threadIdx.x;
  if (idx >= E2 * NHEADS) return;
  int h = idx & 3, e = idx >> 2;
  int s, d;
  if (e < NEDGE) { s = src[e]; d = dst[e]; } else { s = d = e - NEDGE; }
  float mx = o2f(smax[(size_t)d * NHEADS + h]);
  float ex = __expf(score[idx] - mx);
  atomicAdd(denom + (size_t)d * NHEADS + h, ex);
  int HC = NHEADS * C;
  const float4* pl = (const float4*)(xl + (size_t)s * HC + h * C);
  float* po = accum + (size_t)d * HC + h * C;
  for (int c4 = 0; c4 < (C >> 2); ++c4) {
    float4 l = pl[c4];
    atomicAdd(po + 4 * c4 + 0, ex * l.x);
    atomicAdd(po + 4 * c4 + 1, ex * l.y);
    atomicAdd(po + 4 * c4 + 2, ex * l.z);
    atomicAdd(po + 4 * c4 + 3, ex * l.w);
  }
}

// ---------- finalize node: head-mean + bias + ReLU ----------
__global__ void node_finalize(const float* __restrict__ accum, const float* __restrict__ denom,
                              const float* __restrict__ bias, float* __restrict__ out, int C) {
  int idx = blockIdx.x * blockDim.x + threadIdx.x;
  if (idx >= N_NODES * C) return;
  int n = idx / C, c = idx % C;
  int HC = NHEADS * C;
  float sum = 0.f;
#pragma unroll
  for (int h = 0; h < NHEADS; ++h)
    sum += accum[(size_t)n * HC + h * C + c] / (denom[(size_t)n * NHEADS + h] + 1e-16f);
  float v = sum * 0.25f + bias[c];
  out[idx] = v > 0.f ? v : 0.f;
}

// ---------- global mean pool (contiguous 512 nodes / graph, 64 ch) ----------
__global__ void pool_mean(const float* __restrict__ h2, float* __restrict__ g) {
  int b = blockIdx.x, c = threadIdx.x;   // 64 threads
  const float* p = h2 + (size_t)b * N_PER * 64 + c;
  float s = 0.f;
  for (int i = 0; i < N_PER; ++i) s += p[(size_t)i * 64];
  g[b * 64 + c] = s * (1.f / N_PER);
}

// ---------- fused FC head + mask ----------
__global__ void head_mlp(const float* __restrict__ h2, const float* __restrict__ g,
                         const float* __restrict__ Wfc1, const float* __restrict__ bfc1,
                         const float* __restrict__ Wfc2, const float* __restrict__ bfc2,
                         const unsigned char* __restrict__ mask, float* __restrict__ out) {
  int bs = blockIdx.x;                  // B*16 blocks, 64 threads
  int b = bs >> 4, s = bs & 15;
  __shared__ float comb[128];
  __shared__ float hid[64];
  int t = threadIdx.x;
  int node = b * N_PER + c_stations[s];
  comb[t]      = h2[(size_t)node * 64 + t];
  comb[64 + t] = g[b * 64 + t];
  __syncthreads();
  float acc = bfc1[t];
  for (int c = 0; c < 128; ++c) acc += comb[c] * Wfc1[c * 64 + t];
  hid[t] = (acc > 0.f ? acc : 0.f) * Wfc2[t];
  __syncthreads();
  for (int off = 32; off > 0; off >>= 1) {
    if (t < off) hid[t] += hid[t + off];
    __syncthreads();
  }
  if (t == 0) {
    float q = hid[0] + bfc2[0];
    out[bs] = mask[bs] ? q : -1e8f;
  }
}

// ================= launcher =================
extern "C" void kernel_launch(void* const* d_in, const int* in_sizes, int n_in,
                              void* d_out, int out_size, void* d_ws, size_t ws_size,
                              hipStream_t stream) {
  const float* x      = (const float*)d_in[0];
  const int*   eidx   = (const int*)d_in[1];
  const float* eattr  = (const float*)d_in[2];
  const unsigned char* amask = (const unsigned char*)d_in[4];
  const float* W1l = (const float*)d_in[5],  *b1l = (const float*)d_in[6];
  const float* W1r = (const float*)d_in[7],  *b1r = (const float*)d_in[8];
  const float* W1e = (const float*)d_in[9],  *att1 = (const float*)d_in[10];
  const float* bias1 = (const float*)d_in[11];
  const float* W2l = (const float*)d_in[12], *b2l = (const float*)d_in[13];
  const float* W2r = (const float*)d_in[14], *b2r = (const float*)d_in[15];
  const float* W2e = (const float*)d_in[16], *att2 = (const float*)d_in[17];
  const float* bias2 = (const float*)d_in[18];
  const float* Wfc1 = (const float*)d_in[19], *bfc1 = (const float*)d_in[20];
  const float* Wfc2 = (const float*)d_in[21], *bfc2 = (const float*)d_in[22];
  float* out = (float*)d_out;

  const int* src = eidx;
  const int* dst = eidx + NEDGE;

  // ---- workspace layout ----
  char* ws = (char*)d_ws;
  size_t off = 0;
  auto alloc = [&](size_t bytes) { size_t o = off; off += (bytes + 255) & ~(size_t)255; return o; };
  unsigned short* xb   = (unsigned short*)(ws + alloc((size_t)N_NODES * 128 * 2));
  unsigned short* w1lt = (unsigned short*)(ws + alloc(128 * 128 * 2));
  unsigned short* w1rt = (unsigned short*)(ws + alloc(128 * 128 * 2));
  unsigned short* w2lt = (unsigned short*)(ws + alloc(32 * 256 * 2));
  unsigned short* w2rt = (unsigned short*)(ws + alloc(32 * 256 * 2));
  float* cnt   = (float*)(ws + alloc((size_t)N_NODES * 4));
  float* lsum  = (float*)(ws + alloc((size_t)N_NODES * 2 * 4));
  float* lattr = (float*)(ws + alloc((size_t)N_NODES * 2 * 4));
  unsigned* smax = (unsigned*)(ws + alloc((size_t)N_NODES * NHEADS * 4));
  float* denom = (float*)(ws + alloc((size_t)N_NODES * NHEADS * 4));
  float* score = (float*)(ws + alloc((size_t)E2 * NHEADS * 4));
  float* bufA  = (float*)(ws + alloc((size_t)N_NODES * 256 * 4));   // xl (both layers)
  float* bufB  = (float*)(ws + alloc((size_t)N_NODES * 256 * 4));   // xr, then accum
  float* h1    = (float*)(ws + alloc((size_t)N_NODES * 32 * 4));
  unsigned short* h1b = (unsigned short*)(ws + alloc((size_t)N_NODES * 32 * 2));
  float* h2    = (float*)(ws + alloc((size_t)N_NODES * 64 * 4));
  float* gpool = (float*)(ws + alloc((size_t)NGRAPH * 64 * 4));
  (void)ws_size; (void)in_sizes; (void)n_in; (void)out_size;

  const int T = 256;
  auto blk = [](long n, int t) { return (int)((n + t - 1) / t); };
  const unsigned NEG_MAX_ORD = 0x00800000u;   // f2o(-FLT_MAX)

  // convert inputs / repack weights to bf16
  cvt_f32_bf16<<<blk((long)N_NODES * 128, T), T, 0, stream>>>(x, xb, N_NODES * 128);
  repack_wt<<<blk(128 * 128, T), T, 0, stream>>>(W1l, w1lt, 128, 128);
  repack_wt<<<blk(128 * 128, T), T, 0, stream>>>(W1r, w1rt, 128, 128);
  repack_wt<<<blk(32 * 256, T), T, 0, stream>>>(W2l, w2lt, 32, 256);
  repack_wt<<<blk(32 * 256, T), T, 0, stream>>>(W2r, w2rt, 32, 256);

  // self-loop attrs
  fill_u32<<<blk(N_NODES, T), T, 0, stream>>>((unsigned*)cnt, 0u, N_NODES);
  fill_u32<<<blk(N_NODES * 2, T), T, 0, stream>>>((unsigned*)lsum, 0u, N_NODES * 2);
  deg_loop_acc<<<blk(NEDGE, T), T, 0, stream>>>(dst, eattr, cnt, lsum);
  loop_fin<<<blk(N_NODES, T), T, 0, stream>>>(cnt, lsum, lattr);

  // ===== layer 1 (C=32, HC=128) =====
  // strips: (M/16) * (Nc/64) waves, 8 waves/block
  gemm_wmma_bf16<<<(N_NODES / 16) * (128 / 64) / 8, 256, 0, stream>>>(xb, w1lt, b1l, bufA, N_NODES, 128, 128);
  gemm_wmma_bf16<<<(N_NODES / 16) * (128 / 64) / 8, 256, 0, stream>>>(xb, w1rt, b1r, bufB, N_NODES, 128, 128);
  fill_u32<<<blk(N_NODES * NHEADS, T), T, 0, stream>>>(smax, NEG_MAX_ORD, N_NODES * NHEADS);
  fill_u32<<<blk(N_NODES * NHEADS, T), T, 0, stream>>>((unsigned*)denom, 0u, N_NODES * NHEADS);
  edge_score_max<<<blk((long)E2 * NHEADS, T), T, 0, stream>>>(src, dst, eattr, lattr, bufA, bufB, W1e, att1, score, smax, 32);
  fill_u32<<<blk((long)N_NODES * 128, T), T, 0, stream>>>((unsigned*)bufB, 0u, N_NODES * 128);
  edge_accum<<<blk((long)E2 * NHEADS, T), T, 0, stream>>>(src, dst, score, smax, bufA, denom, bufB, 32);
  node_finalize<<<blk((long)N_NODES * 32, T), T, 0, stream>>>(bufB, denom, bias1, h1, 32);
  cvt_f32_bf16<<<blk((long)N_NODES * 32, T), T, 0, stream>>>(h1, h1b, N_NODES * 32);

  // ===== layer 2 (C=64, HC=256) =====
  gemm_wmma_bf16<<<(N_NODES / 16) * (256 / 64) / 8, 256, 0, stream>>>(h1b, w2lt, b2l, bufA, N_NODES, 32, 256);
  gemm_wmma_bf16<<<(N_NODES / 16) * (256 / 64) / 8, 256, 0, stream>>>(h1b, w2rt, b2r, bufB, N_NODES, 32, 256);
  fill_u32<<<blk(N_NODES * NHEADS, T), T, 0, stream>>>(smax, NEG_MAX_ORD, N_NODES * NHEADS);
  fill_u32<<<blk(N_NODES * NHEADS, T), T, 0, stream>>>((unsigned*)denom, 0u, N_NODES * NHEADS);
  edge_score_max<<<blk((long)E2 * NHEADS, T), T, 0, stream>>>(src, dst, eattr, lattr, bufA, bufB, W2e, att2, score, smax, 64);
  fill_u32<<<blk((long)N_NODES * 256, T), T, 0, stream>>>((unsigned*)bufB, 0u, N_NODES * 256);
  edge_accum<<<blk((long)E2 * NHEADS, T), T, 0, stream>>>(src, dst, score, smax, bufA, denom, bufB, 64);
  node_finalize<<<blk((long)N_NODES * 64, T), T, 0, stream>>>(bufB, denom, bias2, h2, 64);

  // ===== pool + head =====
  pool_mean<<<NGRAPH, 64, 0, stream>>>(h2, gpool);
  head_mlp<<<NGRAPH * 16, 64, 0, stream>>>(h2, gpool, Wfc1, bfc1, Wfc2, bfc2, amask, out);
}
